// AttEdgeConv_65000035058607
// MI455X (gfx1250) — matrix operand order repaired and checked by reading, not compiled, not run on hardware
//
#include <hip/hip_runtime.h>
#include <cstdint>

typedef __attribute__((ext_vector_type(16))) _Float16     v16h;
typedef __attribute__((ext_vector_type(8)))  float        v8f;
typedef __attribute__((ext_vector_type(4)))  unsigned int v4u;
typedef __attribute__((ext_vector_type(8)))  int          v8i;
typedef __attribute__((ext_vector_type(4)))  int          v4i;

#define BATCH 16
#define NPTS  2048
#define CH    64
#define KNN   20

// ---------------------------------------------------------------- WMMA helpers
__device__ __forceinline__ v8f wmma16(v16h a, v16h b, v8f c) {
  // D = A(16x32,f16) * B(32x16,f16) + C(16x16,f32)
  return __builtin_amdgcn_wmma_f32_16x16x32_f16(false, a, false, b, (short)0, c,
                                                false, false);
}

// A fragment (16x32 f16, ISA 7.12.2 layout): element (m,k) at base[m*ms + k]
__device__ __forceinline__ v16h load_a(const _Float16* base, int ms, int lane) {
  int m  = lane & 15;
  int kb = (lane >> 4) << 3;   // 0 or 8
  v16h a;
#pragma unroll
  for (int v = 0; v < 8; ++v) {
    int k = (v < 4) ? (kb + 2 * v) : (16 + kb + 2 * (v - 4));
    a[2 * v]     = base[m * ms + k];
    a[2 * v + 1] = base[m * ms + k + 1];
  }
  return a;
}

// B fragment (32x16 f16): element (k,n) at base[k*ks + n*ns]
__device__ __forceinline__ v16h load_b(const _Float16* base, int ks, int ns, int lane) {
  int n  = lane & 15;
  int kb = (lane >> 4) << 3;   // 0 or 8
  v16h b;
#pragma unroll
  for (int v = 0; v < 8; ++v) {
    int k = (v < 4) ? (kb + 2 * v) : (16 + kb + 2 * (v - 4));
    b[2 * v]     = base[k * ks + n * ns];
    b[2 * v + 1] = base[(k + 1) * ks + n * ns];
  }
  return b;
}

// ---------------------------------------------------------------- prep kernels
__global__ void prep_x_kernel(const float* __restrict__ x,
                              _Float16* __restrict__ xh,
                              float* __restrict__ sq) {
  int p = blockIdx.x * blockDim.x + threadIdx.x;
  if (p >= BATCH * NPTS) return;
  const float* src = x + (size_t)p * CH;
  _Float16* dst = xh + (size_t)p * CH;
  float s = 0.0f;
#pragma unroll
  for (int c = 0; c < CH; ++c) {
    float v = src[c];
    s += v * v;
    dst[c] = (_Float16)v;
  }
  sq[p] = s;
}

__global__ void prep_w_kernel(const float* __restrict__ W1,
                              const float* __restrict__ W2,
                              const float* __restrict__ W3,
                              _Float16* __restrict__ wh) {
  for (int i = threadIdx.x; i < 16384; i += blockDim.x) {
    float v;
    if (i < 8192)       v = W1[i];
    else if (i < 12288) v = W2[i - 8192];
    else                v = W3[i - 12288];
    wh[i] = (_Float16)v;
  }
}

// ---------------------------------------------------------------- KNN (WMMA + TDM)
// 256-thread block (8 waves) handles 128 query rows of one batch.
// The 128-point j-panel (16 KB) is staged into LDS by the Tensor Data Mover,
// then every wave computes its 16x128 distance tile from LDS with WMMA.
__global__ void knn_kernel(const _Float16* __restrict__ xh,
                           const float* __restrict__ sq,
                           int* __restrict__ knn_idx) {
  int tid  = threadIdx.x;
  int lane = tid & 31;
  int wave = tid >> 5;             // 0..7
  int bid  = blockIdx.x;
  int b    = bid >> 4;             // NPTS/128 = 16 row-groups per batch
  int i0   = (bid & 15) << 7;      // first of 128 query rows
  int iw   = i0 + wave * 16;       // this wave's 16 rows

  const _Float16* xb  = xh + (size_t)(b * NPTS) * CH;
  const float*    sqb = sq + b * NPTS;

  __shared__ _Float16 panel[128][CH];   // 16 KB: current j-panel (row-major)
  __shared__ float    dt[8][16][16];    // per-wave distance tiles
  __shared__ float    sqA_s[128];
  __shared__ float    sqj_s[128];

  if (tid < 128) sqA_s[tid] = sqb[i0 + tid];

  // A fragments for this wave's 16 query rows (K = channel)
  v16h a0 = load_a(xb + (size_t)iw * CH,      CH, lane);   // K 0..31
  v16h a1 = load_a(xb + (size_t)iw * CH + 32, CH, lane);   // K 32..63

  float kd[KNN]; int ki[KNN];
#pragma unroll
  for (int t = 0; t < KNN; ++t) { kd[t] = 3.0e38f; ki[t] = 0; }
  float worst = 3.0e38f; int wp = 0;

  unsigned int lds_base = (unsigned int)(uintptr_t)(&panel[0][0]);

  __syncthreads();

  for (int j0 = 0; j0 < NPTS; j0 += 128) {
#if __has_builtin(__builtin_amdgcn_tensor_load_to_lds)
    if (wave == 0) {
      // D# descriptor: 2D tile, 8-byte elements.
      // Row = one point (64 f16 = 16 x 8B), 128 rows per tile.
      unsigned long long gaddr =
          (unsigned long long)(uintptr_t)(xb + (size_t)j0 * CH);
      v4u g0;
      g0[0] = 1u;                                    // count = 1, user mode
      g0[1] = lds_base;                              // lds_addr
      g0[2] = (unsigned int)(gaddr & 0xffffffffull); // global_addr[31:0]
      g0[3] = (unsigned int)((gaddr >> 32) & 0x01ffffffull) | (2u << 30); // +type=2
      v8i g1;
      g1[0] = (int)(3u << 16);          // data_size = 3 (8 bytes)
      g1[1] = (int)(16u << 16);         // tensor_dim0 = 16 (low 16 bits)
      g1[2] = (int)(2048u << 16);       // tensor_dim0 hi=0 | tensor_dim1 lo = 2048
      g1[3] = (int)(16u << 16);         // tensor_dim1 hi=0 | tile_dim0 = 16
      g1[4] = 128;                      // tile_dim1 = 128, tile_dim2 = 0
      g1[5] = 16;                       // tensor_dim0_stride = 16 (low 32)
      g1[6] = 0;                        // stride hi | tensor_dim1_stride lo
      g1[7] = 0;
      v4i g2 = {0, 0, 0, 0};
      v4i g3 = {0, 0, 0, 0};
      v8i g4 = {0, 0, 0, 0, 0, 0, 0, 0};
      __builtin_amdgcn_tensor_load_to_lds(g0, g1, g2, g3, g4, 0);
      __builtin_amdgcn_s_wait_tensorcnt(0);
    }
#else
    for (int e = tid; e < 128 * CH; e += 256) {
      int r = e >> 6, c = e & 63;
      panel[r][c] = xb[(size_t)(j0 + r) * CH + c];
    }
#endif
    if (tid < 128) sqj_s[tid] = sqb[j0 + tid];
    __syncthreads();

    for (int jt = 0; jt < 8; ++jt) {
      v16h b0 = load_b(&panel[jt * 16][0],  1, CH, lane);
      v16h b1 = load_b(&panel[jt * 16][32], 1, CH, lane);
      v8f acc = {};
      acc = wmma16(a0, b0, acc);
      acc = wmma16(a1, b1, acc);

      int col   = lane & 15;
      int rbase = (lane >> 4) << 3;
      float sqjv = sqj_s[jt * 16 + col];
#pragma unroll
      for (int r = 0; r < 8; ++r)
        dt[wave][rbase + r][col] =
            sqA_s[wave * 16 + rbase + r] + sqjv - 2.0f * acc[r];

      // wave-private tile: DS ops of one wave complete in order; just stop
      // the compiler from reordering around the cross-lane LDS handoff.
      __builtin_amdgcn_wave_barrier();

      if (lane < 16) {
#pragma unroll
        for (int c = 0; c < 16; ++c) {
          float d = dt[wave][lane][c];
          if (d < worst) {               // replace current worst of top-K
            kd[wp] = d; ki[wp] = j0 + jt * 16 + c;
            worst = kd[0]; wp = 0;
#pragma unroll
            for (int t = 1; t < KNN; ++t)
              if (kd[t] > worst) { worst = kd[t]; wp = t; }
          }
        }
      }
      __builtin_amdgcn_wave_barrier();
    }
    __syncthreads();   // everyone done with panel before next TDM overwrite
  }

  if (lane < 16) {
    int* dst = knn_idx + (size_t)(b * NPTS + iw + lane) * KNN;
#pragma unroll
    for (int t = 0; t < KNN; ++t) dst[t] = ki[t];
  }
}

// ---------------------------------------------------------------- Edge MLP (WMMA)
// One block (4 waves) per point: edge(32x128) -> h1(32x64) -> h2(32x64) -> out(32x64)
// then max over the 20 valid neighbor rows.
__global__ void mlp_kernel(const float* __restrict__ x,
                           const int* __restrict__ knn_idx,
                           const _Float16* __restrict__ w1h,
                           const _Float16* __restrict__ w2h,
                           const _Float16* __restrict__ w3h,
                           const float* __restrict__ b1,
                           const float* __restrict__ b2,
                           const float* __restrict__ b3,
                           float* __restrict__ agg) {
  int p    = blockIdx.x;
  int b    = p >> 11;
  int tid  = threadIdx.x;     // 128 threads = 4 waves
  int lane = tid & 31;
  int ct   = tid >> 5;        // wave id == column tile 0..3

  __shared__ _Float16 eh[32][128];
  __shared__ _Float16 h1[32][64];
  __shared__ float    out32[32][64];

  const float* xi = x + (size_t)p * CH;
  const int*   nb = knn_idx + (size_t)p * KNN;

  // build f16 edge matrix [xi | xj - xi], rows >= KNN zero padded
  for (int e = tid; e < 32 * 128; e += 128) {
    int row = e >> 7, col = e & 127;
    float v = 0.0f;
    if (row < KNN) {
      if (col < CH) v = xi[col];
      else {
        int j = nb[row];
        v = x[((size_t)(b * NPTS) + j) * CH + (col - CH)] - xi[col - CH];
      }
    }
    eh[row][col] = (_Float16)v;
  }
  __syncthreads();

  // layer 1: (32x128) @ W1(128x64), relu
#pragma unroll
  for (int rt = 0; rt < 2; ++rt) {
    v8f acc = {};
#pragma unroll
    for (int kk = 0; kk < 4; ++kk) {
      v16h a  = load_a(&eh[rt * 16][kk * 32], 128, lane);
      v16h bb = load_b(w1h + (size_t)(kk * 32) * 64 + ct * 16, 64, 1, lane);
      acc = wmma16(a, bb, acc);
    }
    int col = ct * 16 + (lane & 15);
    int rbase = (lane >> 4) << 3;
    float bias = b1[col];
#pragma unroll
    for (int r = 0; r < 8; ++r)
      h1[rt * 16 + rbase + r][col] = (_Float16)fmaxf(acc[r] + bias, 0.0f);
  }
  __syncthreads();

  // layer 2: h1 @ W2, relu (output reuses eh storage, stride 64)
  _Float16* h2 = &eh[0][0];
#pragma unroll
  for (int rt = 0; rt < 2; ++rt) {
    v8f acc = {};
#pragma unroll
    for (int kk = 0; kk < 2; ++kk) {
      v16h a  = load_a(&h1[rt * 16][kk * 32], 64, lane);
      v16h bb = load_b(w2h + (size_t)(kk * 32) * 64 + ct * 16, 64, 1, lane);
      acc = wmma16(a, bb, acc);
    }
    int col = ct * 16 + (lane & 15);
    int rbase = (lane >> 4) << 3;
    float bias = b2[col];
#pragma unroll
    for (int r = 0; r < 8; ++r)
      h2[(rt * 16 + rbase + r) * 64 + col] = (_Float16)fmaxf(acc[r] + bias, 0.0f);
  }
  __syncthreads();

  // layer 3: h2 @ W3 (+b3), f32 result
#pragma unroll
  for (int rt = 0; rt < 2; ++rt) {
    v8f acc = {};
#pragma unroll
    for (int kk = 0; kk < 2; ++kk) {
      v16h a  = load_a(h2 + (size_t)(rt * 16) * 64 + kk * 32, 64, lane);
      v16h bb = load_b(w3h + (size_t)(kk * 32) * 64 + ct * 16, 64, 1, lane);
      acc = wmma16(a, bb, acc);
    }
    int col = ct * 16 + (lane & 15);
    int rbase = (lane >> 4) << 3;
    float bias = b3[col];
#pragma unroll
    for (int r = 0; r < 8; ++r)
      out32[rt * 16 + rbase + r][col] = acc[r] + bias;
  }
  __syncthreads();

  // max over the K neighbor rows
  if (tid < CH) {
    float m = -3.0e38f;
#pragma unroll
    for (int r = 0; r < KNN; ++r) m = fmaxf(m, out32[r][tid]);
    agg[(size_t)p * CH + tid] = m;
  }
}

// ---------------------------------------------------------------- SE block
__global__ void se_kernel(const float* __restrict__ agg,
                          const float* __restrict__ se_w1,
                          const float* __restrict__ se_w2,
                          float* __restrict__ y) {
  int b = blockIdx.x;          // 64 threads, thread = channel
  int c = threadIdx.x;
  float s = 0.0f;
  for (int n = 0; n < NPTS; ++n) s += agg[((size_t)(b * NPTS) + n) * CH + c];
  s *= (1.0f / (float)NPTS);

  __shared__ float sl[CH];
  __shared__ float hid[4];
  sl[c] = s;
  __syncthreads();
  if (c < 4) {
    float h = 0.0f;
    for (int k = 0; k < CH; ++k) h += sl[k] * se_w1[k * 4 + c];
    hid[c] = fmaxf(h, 0.0f);
  }
  __syncthreads();
  float v = 0.0f;
#pragma unroll
  for (int t = 0; t < 4; ++t) v += hid[t] * se_w2[t * CH + c];
  y[b * CH + c] = 1.0f / (1.0f + expf(-v));
}

// ---------------------------------------------------------------- channel pooling
__global__ void pool_kernel(const float* __restrict__ agg,
                            const float* __restrict__ y,
                            float* __restrict__ avg_o,
                            float* __restrict__ max_o) {
  int p = blockIdx.x * blockDim.x + threadIdx.x;   // point index over B*N
  if (p >= BATCH * NPTS) return;
  int b = p >> 11;
  const float* yb = y + b * CH;
  float s = 0.0f, m = -3.0e38f;
#pragma unroll
  for (int c = 0; c < CH; ++c) {
    float v = agg[(size_t)p * CH + c] * yb[c];
    s += v;
    m = fmaxf(m, v);
  }
  avg_o[p] = s * (1.0f / (float)CH);
  max_o[p] = m;
}

// ---------------------------------------------------------------- conv7 + residual
__global__ void final_kernel(const float* __restrict__ x,
                             const float* __restrict__ agg,
                             const float* __restrict__ y,
                             const float* __restrict__ avg_o,
                             const float* __restrict__ max_o,
                             const float* __restrict__ sa_w,
                             float* __restrict__ out) {
  int g = blockIdx.x * blockDim.x + threadIdx.x;   // B*N*C elements
  if (g >= BATCH * NPTS * CH) return;
  int p = g >> 6;
  int c = g & 63;
  int b = p >> 11;
  int n = p & (NPTS - 1);

  float a = 0.0f;
#pragma unroll
  for (int t = 0; t < 7; ++t) {
    int nn = n + t - 3;
    if (nn >= 0 && nn < NPTS) {
      int q = b * NPTS + nn;
      a += avg_o[q] * sa_w[t] + max_o[q] * sa_w[7 + t];
    }
  }
  float att = 1.0f / (1.0f + expf(-a));
  out[g] = agg[g] * y[b * CH + c] * att + x[g];
}

// ---------------------------------------------------------------- launch
extern "C" void kernel_launch(void* const* d_in, const int* in_sizes, int n_in,
                              void* d_out, int out_size, void* d_ws, size_t ws_size,
                              hipStream_t stream) {
  (void)in_sizes; (void)n_in; (void)out_size; (void)ws_size;
  const float* x     = (const float*)d_in[0];
  const float* W1    = (const float*)d_in[3];
  const float* b1    = (const float*)d_in[4];
  const float* W2    = (const float*)d_in[5];
  const float* b2    = (const float*)d_in[6];
  const float* W3    = (const float*)d_in[7];
  const float* b3    = (const float*)d_in[8];
  const float* se_w1 = (const float*)d_in[9];
  const float* se_w2 = (const float*)d_in[10];
  const float* sa_w  = (const float*)d_in[11];

  char* ws = (char*)d_ws;
  size_t off = 0;
  auto carve = [&](size_t bytes) {
    void* p = ws + off;
    off += (bytes + 255) & ~(size_t)255;
    return p;
  };
  _Float16* xh      = (_Float16*)carve((size_t)BATCH * NPTS * CH * 2);     // 4 MB
  _Float16* wh      = (_Float16*)carve(16384 * 2);                         // 32 KB
  float*    sq      = (float*)   carve((size_t)BATCH * NPTS * 4);          // 128 KB
  int*      knn_idx = (int*)     carve((size_t)BATCH * NPTS * KNN * 4);    // 2.5 MB
  float*    agg     = (float*)   carve((size_t)BATCH * NPTS * CH * 4);     // 8 MB
  float*    yv      = (float*)   carve((size_t)BATCH * CH * 4);
  float*    avg_o   = (float*)   carve((size_t)BATCH * NPTS * 4);
  float*    max_o   = (float*)   carve((size_t)BATCH * NPTS * 4);

  prep_x_kernel<<<(BATCH * NPTS + 255) / 256, 256, 0, stream>>>(x, xh, sq);
  prep_w_kernel<<<1, 256, 0, stream>>>(W1, W2, W3, wh);
  knn_kernel<<<BATCH * (NPTS / 128), 256, 0, stream>>>(xh, sq, knn_idx);
  mlp_kernel<<<BATCH * NPTS, 128, 0, stream>>>(x, knn_idx,
                                               wh, wh + 8192, wh + 12288,
                                               b1, b2, b3, agg);
  se_kernel<<<BATCH, CH, 0, stream>>>(agg, se_w1, se_w2, yv);
  pool_kernel<<<(BATCH * NPTS + 255) / 256, 256, 0, stream>>>(agg, yv, avg_o, max_o);
  final_kernel<<<(BATCH * NPTS * CH + 255) / 256, 256, 0, stream>>>(
      x, agg, yv, avg_o, max_o, sa_w, (float*)d_out);
}
